// Decoder_6640019440041
// MI455X (gfx1250) — compile-verified, matrix operand order checked by reading, not run on hardware
//
#include <hip/hip_runtime.h>

// ---------------------------------------------------------------------------
// Decoder (LSTM-like scan with attention-ish gating) for MI455X / gfx1250.
//
// - All GEMMs via v_wmma_f32_16x16x32_bf16 (f32 accumulate), weights converted
//   to bf16 once per launch; ~17MB of bf16 scan weights stay L2-resident (192MB) across
//   the 96 sequential scan steps.
// - Input projection (y@w_ih^T + b_ih + b_hh) and h_last@fc1[:,H:]^T hoisted
//   out of the scan as large parallel GEMMs.
// - GEMM kernel: 128x128 block tile, K-tile 64, 8 wave32 waves (2Mx4N), each
//   wave 4x2 WMMA accumulators. Double-buffered LDS fed by CDNA5 async
//   global->LDS DMA (global_load_async_to_lds_b128, ASYNCcnt): zero VGPR
//   staging, so no scratch spills; one s_wait_asynccnt + barrier per K-tile
//   and the copy runs fully under the 16 WMMAs.
// ---------------------------------------------------------------------------

#define BQ 256
#define TQ 96
#define HQ 1024
#define FQ 512
#define GQ 4096   // 4*H
#define K2 2048   // 2*H

typedef __attribute__((ext_vector_type(16))) __bf16 v16bf;
typedef __attribute__((ext_vector_type(8)))  float  v8f;

// ---- bf16 helpers (raw bit storage as unsigned short) ----------------------
__device__ __forceinline__ unsigned short f2bf(float f) {
    unsigned int u = __float_as_uint(f);
    u += 0x7FFFu + ((u >> 16) & 1u);   // round-to-nearest-even
    return (unsigned short)(u >> 16);
}
__device__ __forceinline__ float bf2f(unsigned short h) {
    return __uint_as_float(((unsigned int)h) << 16);
}
__device__ __forceinline__ float sigmoidf(float x) {
    return 1.0f / (1.0f + __expf(-x));
}

// ---- CDNA5 async global->LDS copy (16B per lane, tracked by ASYNCcnt) ------
// vdst holds the per-lane LDS byte address; low 32 bits of a generic pointer
// to LDS are the wave-relative LDS offset.
__device__ __forceinline__ void async_copy_b128(const unsigned short* gsrc,
                                                unsigned short* ldst) {
    unsigned lds_addr = (unsigned)(unsigned long long)ldst;
    asm volatile("global_load_async_to_lds_b128 %0, %1, off"
                 :: "v"(lds_addr), "v"(gsrc)
                 : "memory");
}
__device__ __forceinline__ void wait_async0() {
    asm volatile("s_wait_asynccnt 0x0" ::: "memory");
}

// ---- tiled WMMA GEMM: D = A(MxK,bf16) * W(NxK,bf16)^T [+bias][+Cf][+Cbf] ---
// M,N multiples of 128; K multiple of 64 (true at every call site).
// LDS rows padded to 72 bf16 (144B): 16B-aligned ds_load_b128 fragment reads,
// conflict-free bank pattern.
#define LDSW 72

union FragU {
    uint4 u[2];
    v16bf v;
};

__device__ __forceinline__ v16bf load_frag(const unsigned short* row, int ks, int khalf) {
    FragU f;
    f.u[0] = *reinterpret_cast<const uint4*>(row + ks + khalf * 8);
    f.u[1] = *reinterpret_cast<const uint4*>(row + ks + 16 + khalf * 8);
    return f.v;
}

__global__ __launch_bounds__(256) void gemm_bf16_wmma(
    const unsigned short* __restrict__ A,  int lda,
    const unsigned short* __restrict__ W,  int ldb,
    const float* __restrict__ bias,                      // len N or null
    const float* __restrict__ Cf,  int ldcf,             // f32 addend matrix or null
    const unsigned short* __restrict__ Cbf, int ldcbf,   // bf16 addend matrix or null
    float* __restrict__ Df,  int lddf,                   // f32 output or null
    unsigned short* __restrict__ Dbf, int lddbf,         // bf16 output or null
    int K)
{
    __shared__ __align__(16) unsigned short As[2][128 * LDSW];
    __shared__ __align__(16) unsigned short Bs[2][128 * LDSW];

    const int tid   = threadIdx.x;
    const int lane  = tid & 31;
    const int wid   = tid >> 5;
    const int waveM = wid & 1;        // 2 waves along M (64 rows each)
    const int waveN = wid >> 1;       // 4 waves along N (32 cols each)
    const int lrow  = lane & 15;
    const int khalf = lane >> 4;

    const size_t mBase = (size_t)blockIdx.y * 128;
    const size_t nBase = (size_t)blockIdx.x * 128;

    v8f acc[4][2];
#pragma unroll
    for (int i = 0; i < 4; ++i)
#pragma unroll
        for (int j = 0; j < 2; ++j)
            acc[i][j] = (v8f){0.f, 0.f, 0.f, 0.f, 0.f, 0.f, 0.f, 0.f};

    // Per-thread staging chunk coordinates: 1024 16B-chunks per tile,
    // 4 per thread. chunk c -> row c>>3, element offset (c&7)*8.
    // Issue async DMA for one K-tile into LDS buffer `buf`.
    auto issue_tile = [&](int buf, int kBase) {
#pragma unroll
        for (int i = 0; i < 4; ++i) {
            const int c   = tid + i * 256;
            const int row = c >> 3;
            const int off = (c & 7) * 8;
            async_copy_b128(A + (mBase + row) * (size_t)lda + kBase + off,
                            &As[buf][row * LDSW + off]);
            async_copy_b128(W + (nBase + row) * (size_t)ldb + kBase + off,
                            &Bs[buf][row * LDSW + off]);
        }
    };

    // Prologue: start DMA of K-tile 0 into buffer 0.
    issue_tile(0, 0);

    int cur = 0;
    for (int k0 = 0; k0 < K; k0 += 64) {
        wait_async0();     // this wave's DMA into `cur` done
        __syncthreads();   // publish buffer `cur`; retire last iter's reads of `cur^1`

        if ((k0 + 64) < K) {
            // DMA next tile into the other buffer; its readers finished
            // before the barrier above, and the copy runs under the WMMAs.
            issue_tile(cur ^ 1, k0 + 64);
        }

#pragma unroll
        for (int ks = 0; ks < 64; ks += 32) {
            v16bf afrag[4];
            v16bf bfrag[2];
#pragma unroll
            for (int ms = 0; ms < 4; ++ms) {
                const int r = waveM * 64 + ms * 16 + lrow;
                afrag[ms] = load_frag(&As[cur][r * LDSW], ks, khalf);
            }
#pragma unroll
            for (int ns = 0; ns < 2; ++ns) {
                const int r = waveN * 32 + ns * 16 + lrow;
                bfrag[ns] = load_frag(&Bs[cur][r * LDSW], ks, khalf);
            }
#pragma unroll
            for (int ms = 0; ms < 4; ++ms)
#pragma unroll
                for (int ns = 0; ns < 2; ++ns)
                    acc[ms][ns] = __builtin_amdgcn_wmma_f32_16x16x32_bf16(
                        false, afrag[ms], false, bfrag[ns],
                        (short)0, acc[ms][ns], false, false);
        }
        cur ^= 1;
    }

    // Epilogue: C/D 16x16 f32 layout => lane holds col n=lane&15, rows
    // m = (lane>>4)*8 + r for r in 0..7 (one row per acc VGPR).
#pragma unroll
    for (int ms = 0; ms < 4; ++ms) {
#pragma unroll
        for (int ns = 0; ns < 2; ++ns) {
            const size_t n = nBase + waveN * 32 + ns * 16 + lrow;
            const float bn = bias ? bias[n] : 0.f;
#pragma unroll
            for (int r = 0; r < 8; ++r) {
                const size_t m = mBase + waveM * 64 + ms * 16 + khalf * 8 + r;
                float v = acc[ms][ns][r] + bn;
                if (Cf)  v += Cf[m * (size_t)ldcf + n];
                if (Cbf) v += bf2f(Cbf[m * (size_t)ldcbf + n]);
                if (Df)  Df[m * (size_t)lddf + n] = v;
                if (Dbf) Dbf[m * (size_t)lddbf + n] = f2bf(v);
            }
        }
    }
}

// ---- small helper kernels --------------------------------------------------
__global__ void f2bf_kernel(const float* __restrict__ s, unsigned short* __restrict__ d, size_t n) {
    size_t i = (size_t)blockIdx.x * blockDim.x + threadIdx.x;
    const size_t stride = (size_t)gridDim.x * blockDim.x;
    for (; i < n; i += stride) d[i] = f2bf(s[i]);
}

__global__ void add_bias_kernel(const float* __restrict__ a, const float* __restrict__ b,
                                float* __restrict__ o, int n) {
    int i = blockIdx.x * blockDim.x + threadIdx.x;
    if (i < n) o[i] = a[i] + b[i];
}

__global__ void zero_state_kernel(float* __restrict__ c, unsigned short* __restrict__ d, int n) {
    int i = blockIdx.x * blockDim.x + threadIdx.x;
    if (i < n) { c[i] = 0.f; d[i] = 0; }
}

// Xcat[b, 0:H] = d[b,:]; Xcat[b, H:2H] = h_enc[b, t-1, :] (zeros at t==0)
__global__ void pack_xcat_kernel(const unsigned short* __restrict__ d,
                                 const unsigned short* __restrict__ henc,
                                 int t, unsigned short* __restrict__ xcat) {
    int i = blockIdx.x * blockDim.x + threadIdx.x;   // over B*2H
    if (i >= BQ * K2) return;
    const int b = i >> 11;
    const int j = i & (K2 - 1);
    unsigned short v;
    if (j < HQ) {
        v = d[b * HQ + j];
    } else {
        v = (t > 0) ? henc[(size_t)b * TQ * HQ + (size_t)(t - 1) * HQ + (j - HQ)]
                    : (unsigned short)0;
    }
    xcat[i] = v;
}

// gates (B x 4H, order i,f,g,o) + c -> c_new (f32), d_new (bf16)
__global__ void lstm_pointwise_kernel(const float* __restrict__ gates,
                                      float* __restrict__ c,
                                      unsigned short* __restrict__ d) {
    int idx = blockIdx.x * blockDim.x + threadIdx.x;   // over B*H
    if (idx >= BQ * HQ) return;
    const int b = idx >> 10;
    const int j = idx & (HQ - 1);
    const float* g = gates + (size_t)b * GQ;
    const float ig = sigmoidf(g[j]);
    const float fg = sigmoidf(g[HQ + j]);
    const float gg = tanhf(g[2 * HQ + j]);
    const float og = sigmoidf(g[3 * HQ + j]);
    const float cn = fg * c[idx] + ig * gg;
    c[idx] = cn;
    d[idx] = f2bf(og * tanhf(cn));
}

// ---------------------------------------------------------------------------
extern "C" void kernel_launch(void* const* d_in, const int* in_sizes, int n_in,
                              void* d_out, int out_size, void* d_ws, size_t ws_size,
                              hipStream_t stream) {
    (void)in_sizes; (void)n_in; (void)out_size; (void)ws_size;

    const float* h       = (const float*)d_in[0];
    const float* y_seq   = (const float*)d_in[1];
    const float* fc0_w   = (const float*)d_in[2];
    const float* fc0_b   = (const float*)d_in[3];
    const float* attn1_w = (const float*)d_in[4];
    const float* attn1_b = (const float*)d_in[5];
    const float* w_ih    = (const float*)d_in[6];
    const float* w_hh    = (const float*)d_in[7];
    const float* b_ih    = (const float*)d_in[8];
    const float* b_hh    = (const float*)d_in[9];
    const float* fc1_w   = (const float*)d_in[10];
    const float* fc1_b   = (const float*)d_in[11];
    const float* fc2_w   = (const float*)d_in[12];
    const float* fc2_b   = (const float*)d_in[13];
    float* out = (float*)d_out;

    // --- workspace carve-up (bytes, 256-aligned) ---
    char* ws = (char*)d_ws;
    size_t off = 0;
    auto alloc = [&](size_t bytes) {
        void* p = ws + off;
        off = (off + bytes + 255) & ~(size_t)255;
        return p;
    };
    unsigned short* h_bf    = (unsigned short*)alloc((size_t)BQ * TQ * FQ * 2);
    unsigned short* y_bf    = (unsigned short*)alloc((size_t)BQ * TQ * FQ * 2);
    unsigned short* fc0_bf  = (unsigned short*)alloc((size_t)HQ * FQ * 2);
    unsigned short* attn_bf = (unsigned short*)alloc((size_t)HQ * K2 * 2);
    unsigned short* wih_bf  = (unsigned short*)alloc((size_t)GQ * FQ * 2);
    unsigned short* whh_bf  = (unsigned short*)alloc((size_t)GQ * HQ * 2);
    unsigned short* fc1_bf  = (unsigned short*)alloc((size_t)HQ * K2 * 2);
    unsigned short* fc2_bf  = (unsigned short*)alloc((size_t)FQ * HQ * 2);
    unsigned short* henc_bf = (unsigned short*)alloc((size_t)BQ * TQ * HQ * 2);
    unsigned short* xg_bf   = (unsigned short*)alloc((size_t)BQ * TQ * GQ * 2);
    float*          bias_g  = (float*)alloc((size_t)GQ * 4);
    float*          hlfc1   = (float*)alloc((size_t)BQ * HQ * 4);
    unsigned short* xcat    = (unsigned short*)alloc((size_t)BQ * K2 * 2);
    unsigned short* h1_bf   = (unsigned short*)alloc((size_t)BQ * HQ * 2);
    float*          gates   = (float*)alloc((size_t)BQ * GQ * 4);
    float*          cstate  = (float*)alloc((size_t)BQ * HQ * 4);
    unsigned short* d_bf    = (unsigned short*)alloc((size_t)BQ * HQ * 2);
    unsigned short* r_bf    = (unsigned short*)alloc((size_t)BQ * HQ * 2);

    // --- conversions f32 -> bf16 ---
    auto conv = [&](const float* s, unsigned short* d, size_t n) {
        int blocks = (int)((n + 255) / 256);
        if (blocks > 4096) blocks = 4096;
        f2bf_kernel<<<blocks, 256, 0, stream>>>(s, d, n);
    };
    conv(h,       h_bf,    (size_t)BQ * TQ * FQ);
    conv(y_seq,   y_bf,    (size_t)BQ * TQ * FQ);
    conv(fc0_w,   fc0_bf,  (size_t)HQ * FQ);
    conv(attn1_w, attn_bf, (size_t)HQ * K2);
    conv(w_ih,    wih_bf,  (size_t)GQ * FQ);
    conv(w_hh,    whh_bf,  (size_t)GQ * HQ);
    conv(fc1_w,   fc1_bf,  (size_t)HQ * K2);
    conv(fc2_w,   fc2_bf,  (size_t)FQ * HQ);

    add_bias_kernel<<<(GQ + 255) / 256, 256, 0, stream>>>(b_ih, b_hh, bias_g, GQ);
    zero_state_kernel<<<(BQ * HQ + 255) / 256, 256, 0, stream>>>(cstate, d_bf, BQ * HQ);

    auto gemm = [&](const unsigned short* A, int lda, const unsigned short* W, int ldb,
                    const float* bias, const float* Cf, int ldcf,
                    const unsigned short* Cbf, int ldcbf,
                    float* Df, int lddf, unsigned short* Dbf, int lddbf,
                    int M, int N, int K) {
        dim3 grid(N / 128, M / 128);
        gemm_bf16_wmma<<<grid, 256, 0, stream>>>(A, lda, W, ldb, bias, Cf, ldcf,
                                                 Cbf, ldcbf, Df, lddf, Dbf, lddbf, K);
    };

    // --- hoisted GEMMs ---
    // h_enc = h @ fc0_w^T + fc0_b   : (B*T, F) x (H, F) -> bf16 (B*T, H)
    gemm(h_bf, FQ, fc0_bf, FQ, fc0_b, nullptr, 0, nullptr, 0,
         nullptr, 0, henc_bf, HQ, BQ * TQ, HQ, FQ);
    // xg = y @ w_ih^T + (b_ih+b_hh) : (B*T, F) x (4H, F) -> bf16 (B*T, 4H)
    gemm(y_bf, FQ, wih_bf, FQ, bias_g, nullptr, 0, nullptr, 0,
         nullptr, 0, xg_bf, GQ, BQ * TQ, GQ, FQ);
    // hl_fc1 = h_last @ fc1_w[:, H:]^T + fc1_b : A = strided slice of h_enc at t=T-1
    gemm(henc_bf + (size_t)(TQ - 1) * HQ, TQ * HQ, fc1_bf + HQ, K2, fc1_b,
         nullptr, 0, nullptr, 0, hlfc1, HQ, nullptr, 0, BQ, HQ, HQ);

    // --- sequential scan over T ---
    for (int t = 0; t < TQ; ++t) {
        // Xcat = [d | h_prev_t]
        pack_xcat_kernel<<<(BQ * K2 + 255) / 256, 256, 0, stream>>>(d_bf, henc_bf, t, xcat);
        // h1 = Xcat @ attn1_w^T + attn1_b            (256 x 2048 x 1024)
        gemm(xcat, K2, attn_bf, K2, attn1_b, nullptr, 0, nullptr, 0,
             nullptr, 0, h1_bf, HQ, BQ, HQ, K2);
        // gates = xg[:, t, :] + h1 @ w_hh^T          (256 x 1024 x 4096)
        gemm(h1_bf, HQ, whh_bf, HQ, nullptr, nullptr, 0,
             xg_bf + (size_t)t * GQ, TQ * GQ, gates, GQ, nullptr, 0, BQ, GQ, HQ);
        // LSTM pointwise -> c, d
        lstm_pointwise_kernel<<<(BQ * HQ + 255) / 256, 256, 0, stream>>>(gates, cstate, d_bf);
        // r = d @ fc1_w[:, :H]^T + hl_fc1            (256 x 1024 x 1024)
        gemm(d_bf, HQ, fc1_bf, K2, nullptr, hlfc1, HQ, nullptr, 0,
             nullptr, 0, r_bf, HQ, BQ, HQ, HQ);
        // out[:, t, :] = r @ fc2_w^T + fc2_b         (256 x 1024 x 512)
        gemm(r_bf, HQ, fc2_bf, HQ, fc2_b, nullptr, 0, nullptr, 0,
             out + (size_t)t * FQ, TQ * FQ, nullptr, 0, BQ, FQ, HQ);
    }
}
